// DefectDetector_58265526337847
// MI455X (gfx1250) — matrix operand order repaired and checked by reading, not compiled, not run on hardware
//
#include <hip/hip_runtime.h>
#include <math.h>

// ---------------------------------------------------------------------------
// CDNA5 WMMA types
// ---------------------------------------------------------------------------
typedef __attribute__((ext_vector_type(16))) __bf16 v16bf;
typedef __attribute__((ext_vector_type(8)))  float  v8f;

#define BT 64           // block tile (M and N)
#define KT 32           // K slab per WMMA step
#define LDPA 36         // A-tile LDS row stride in fp32  (144 B, 16B aligned)
#define LDPB 40         // B-tile LDS row stride in bf16  (80 B, 16B aligned)

// A-frag from fp32 LDS row, converting to bf16.
// 16-bit A(16x32): lane half kf=(lane>>4)*16 -> K = {kf..kf+15} (contiguous, matches B)
__device__ inline v16bf ld_frag_a32(const float* row, int kf) {
    v16bf r;
    #pragma unroll
    for (int e = 0; e < 8; ++e) r[e]     = (__bf16)row[kf + e];
    #pragma unroll
    for (int e = 0; e < 8; ++e) r[e + 8] = (__bf16)row[kf + 8 + e];
    return r;
}
// B-frag 16-bit B(32x16): 16 contiguous K bf16 (same k order as A) -> two ds_load_b128
__device__ inline v16bf ld_frag_b(const __bf16* row, int kf) {
    union { uint4 q[2]; v16bf v; } u;
    u.q[0] = *(const uint4*)(row + kf);
    u.q[1] = *(const uint4*)(row + kf + 8);
    return u.v;
}

// ---------------------------------------------------------------------------
// Implicit-GEMM convolution, bf16 WMMA, fused bias/BN/activation epilogue.
//   A = weights [Cout, K] staged fp32 in LDS via GLOBAL_LOAD_ASYNC_TO_LDS_B128
//   B = im2col(x) [K, N*OH*OW], staged transposed [n][k] as bf16
// Block: 128 threads = 4 waves (2x2), 64x64 tile; each wave: 32x32 sub-tile,
// 4 accumulators, 4 v_wmma per K-step. act: 0=none, 1=relu, 2=sigmoid
// ---------------------------------------------------------------------------
template <int KH, int KW>
__global__ __launch_bounds__(128)
void conv_wmma_kernel(const float* __restrict__ x, const float* __restrict__ w,
                      const float* __restrict__ bias,
                      const float* __restrict__ bscale,
                      const float* __restrict__ bshift,
                      float* __restrict__ y,
                      int N, int Cin, int IH, int IW,
                      int Cout, int OH, int OW,
                      int stride, int pad, int act)
{
    constexpr int KHW = KH * KW;
    const int K    = Cin * KHW;
    const int OHW  = OH * OW;
    const int NOut = N * OHW;

    const int Mbase = blockIdx.y * BT;
    const int Nbase = blockIdx.x * BT;

    __shared__ float  As32[BT][LDPA];   // weights [m][k], fp32 (async target)
    __shared__ __bf16 Bs[BT][LDPB];     // im2col  [n][k], bf16 (transposed)

    const int tid   = threadIdx.x;
    const int lane  = tid & 31;
    const int wave  = tid >> 5;
    const int waveM = wave >> 1;        // 0..1
    const int waveN = wave & 1;         // 0..1

    // staging: each thread owns one LDS row half: row = tid>>1, 16 contiguous k
    const int srow = tid >> 1;
    const int koff = (tid & 1) * 16;

    // hoisted im2col column decomposition (loop-invariant)
    const int nB = Nbase + srow;
    const bool nValid = nB < NOut;
    int iy0 = 0, ix0 = 0, bB = 0;
    if (nValid) {
        bB = nB / OHW;
        int sp = nB - bB * OHW;
        int oy = sp / OW, ox = sp - oy * OW;
        iy0 = oy * stride - pad;
        ix0 = ox * stride - pad;
    }
    const int   mA = Mbase + srow;
    const bool  mValid = mA < Cout;
    const float* wrow = w + (size_t)mA * K;
    const float* xcol = x + (size_t)bB * Cin * IH * IW;

    // LDS byte offset of this thread's A staging chunk (generic-addr low 32
    // bits == LDS offset on gfx1250)
    const unsigned ldsA = (unsigned)(size_t)&As32[srow][koff];

    v8f acc00 = {0.f,0.f,0.f,0.f,0.f,0.f,0.f,0.f};
    v8f acc01 = acc00, acc10 = acc00, acc11 = acc00;

    for (int kk = 0; kk < K; kk += KT) {
        // ---- stage A: async global->LDS DMA of 16 contiguous fp32 ----
        if (mValid && (kk + koff + 16 <= K)) {
            const float* gp = wrow + kk + koff;
            #pragma unroll
            for (int q = 0; q < 4; ++q) {
                unsigned long long ga = (unsigned long long)(size_t)(gp + q * 4);
                asm volatile("global_load_async_to_lds_b128 %0, %1, off"
                             :: "v"(ldsA + q * 16), "v"(ga) : "memory");
            }
        } else {
            #pragma unroll
            for (int j = 0; j < 16; ++j) {
                int k = kk + koff + j;
                As32[srow][koff + j] = (mValid && k < K) ? wrow[k] : 0.f;
            }
        }

        // ---- stage B: im2col gather, constant-divisor decomposition ----
        #pragma unroll
        for (int j = 0; j < 16; ++j) {
            int k = kk + koff + j;
            float v = 0.f;
            if (nValid && k < K) {
                int ci  = k / KHW;                 // KHW is constexpr
                int rem = k - ci * KHW;
                int fy  = rem / KW;                // KW is constexpr
                int fx  = rem - fy * KW;
                int iy  = iy0 + fy, ix = ix0 + fx;
                if (iy >= 0 && iy < IH && ix >= 0 && ix < IW)
                    v = xcol[((size_t)ci * IH + iy) * IW + ix];
            }
            Bs[srow][koff + j] = (__bf16)v;
        }

        // async A-tile transfers must land in LDS before the barrier
        asm volatile("s_wait_asynccnt 0" ::: "memory");
        __syncthreads();

        // ---- fragments ----
        const int kfA = (lane >> 4) * 16;
        const int kfB = (lane >> 4) * 16;
        const int mf  = waveM * 32 + (lane & 15);
        const int nf  = waveN * 32 + (lane & 15);

        v16bf a0 = ld_frag_a32(&As32[mf][0],      kfA);
        v16bf a1 = ld_frag_a32(&As32[mf + 16][0], kfA);
        v16bf b0 = ld_frag_b(&Bs[nf][0],          kfB);
        v16bf b1 = ld_frag_b(&Bs[nf + 16][0],     kfB);

        acc00 = __builtin_amdgcn_wmma_f32_16x16x32_bf16(false, a0, false, b0, (short)0, acc00, false, false);
        acc01 = __builtin_amdgcn_wmma_f32_16x16x32_bf16(false, a0, false, b1, (short)0, acc01, false, false);
        acc10 = __builtin_amdgcn_wmma_f32_16x16x32_bf16(false, a1, false, b0, (short)0, acc10, false, false);
        acc11 = __builtin_amdgcn_wmma_f32_16x16x32_bf16(false, a1, false, b1, (short)0, acc11, false, false);
        __syncthreads();
    }

    // ---- epilogue: C layout VGPR r -> M = r + 8*(lane>=16), N = lane&15 ----
    const int mrow0 = Mbase + waveM * 32 + ((lane >> 4) * 8);
    const int ncol0 = Nbase + waveN * 32 + (lane & 15);

    #pragma unroll
    for (int nt = 0; nt < 2; ++nt) {
        int ncol = ncol0 + nt * 16;
        if (ncol >= NOut) continue;
        int b  = ncol / OHW;
        int sp = ncol - b * OHW;
        #pragma unroll
        for (int mt = 0; mt < 2; ++mt) {
            v8f acc = (nt == 0) ? (mt == 0 ? acc00 : acc10)
                                : (mt == 0 ? acc01 : acc11);
            #pragma unroll
            for (int r = 0; r < 8; ++r) {
                int m = mrow0 + mt * 16 + r;
                if (m < Cout) {
                    float v = acc[r];
                    if (bias)   v += bias[m];
                    if (bscale) v = v * bscale[m] + bshift[m];
                    if (act == 1)      v = fmaxf(v, 0.f);
                    else if (act == 2) v = 1.f / (1.f + __expf(-v));
                    y[(size_t)(b * Cout + m) * OHW + sp] = v;
                }
            }
        }
    }
}

// ---------------------------------------------------------------------------
// MaxPool2d(kernel=3, stride=2, padding=1), pad value -inf
// ---------------------------------------------------------------------------
__global__ void maxpool3x3s2_kernel(const float* __restrict__ in,
                                    float* __restrict__ out,
                                    int NC, int IH, int IW, int OH, int OW)
{
    int idx   = blockIdx.x * blockDim.x + threadIdx.x;
    int total = NC * OH * OW;
    if (idx >= total) return;
    int nc = idx / (OH * OW);
    int sp = idx - nc * (OH * OW);
    int oy = sp / OW, ox = sp - oy * OW;
    const float* p = in + (size_t)nc * IH * IW;
    float m = -INFINITY;
    #pragma unroll
    for (int dy = 0; dy < 3; ++dy) {
        int iy = oy * 2 - 1 + dy;
        if (iy < 0 || iy >= IH) continue;
        #pragma unroll
        for (int dx = 0; dx < 3; ++dx) {
            int ix = ox * 2 - 1 + dx;
            if (ix < 0 || ix >= IW) continue;
            m = fmaxf(m, p[iy * IW + ix]);
        }
    }
    out[idx] = m;
}

// ---------------------------------------------------------------------------
// Global average pool: one block per (n,c)
// ---------------------------------------------------------------------------
__global__ void gap_kernel(const float* __restrict__ feat,
                           float* __restrict__ out, int HW)
{
    __shared__ float red[256];
    int bc = blockIdx.x;
    const float* p = feat + (size_t)bc * HW;
    float s = 0.f;
    for (int i = threadIdx.x; i < HW; i += blockDim.x) s += p[i];
    red[threadIdx.x] = s;
    __syncthreads();
    for (int off = blockDim.x >> 1; off > 0; off >>= 1) {
        if ((int)threadIdx.x < off) red[threadIdx.x] += red[threadIdx.x + off];
        __syncthreads();
    }
    if (threadIdx.x == 0) out[bc] = red[0] / (float)HW;
}

// ---------------------------------------------------------------------------
// Fully-connected: out[n][j] = act(sum_k in[n][k]*W[j][k] + b[j])
// ---------------------------------------------------------------------------
__global__ void fc_kernel(const float* __restrict__ in,
                          const float* __restrict__ W,
                          const float* __restrict__ bias,
                          float* __restrict__ out,
                          int Nb, int K, int J, int relu_on)
{
    int idx = blockIdx.x * blockDim.x + threadIdx.x;
    if (idx >= Nb * J) return;
    int n = idx / J, j = idx - n * J;
    const float* a  = in + (size_t)n * K;
    const float* ww = W + (size_t)j * K;
    float s = bias[j];
    for (int k = 0; k < K; ++k) s += a[k] * ww[k];
    if (relu_on) s = fmaxf(s, 0.f);
    out[idx] = s;
}

// ---------------------------------------------------------------------------
// Decode: softmax-4, box decode + clip, scores, mask, labels
// ---------------------------------------------------------------------------
__global__ void decode_kernel(const float* __restrict__ bbox,
                              const float* __restrict__ obj,
                              const float* __restrict__ logits,
                              float* __restrict__ boxes,
                              float* __restrict__ scores,
                              float* __restrict__ labels,
                              float* __restrict__ maskf,
                              int B, int H, int W, float img, float stride)
{
    int idx = blockIdx.x * blockDim.x + threadIdx.x;
    int HW = H * W;
    if (idx >= B * HW) return;
    int b  = idx / HW;
    int sp = idx - b * HW;
    int yy = sp / W, xx = sp - yy * W;

    const float* lg = logits + b * 4;
    float mx = fmaxf(fmaxf(lg[0], lg[1]), fmaxf(lg[2], lg[3]));
    float e0 = __expf(lg[0] - mx), e1 = __expf(lg[1] - mx);
    float e2 = __expf(lg[2] - mx), e3 = __expf(lg[3] - mx);
    float inv = 1.f / (e0 + e1 + e2 + e3);
    float p1 = e1 * inv, p2 = e2 * inv, p3 = e3 * inv;
    float maxc  = fmaxf(p1, fmaxf(p2, p3));
    int   label = (p1 >= p2 && p1 >= p3) ? 1 : ((p2 >= p3) ? 2 : 3);

    const float* bb = bbox + (size_t)b * 4 * HW + sp;
    float d0 = bb[0], d1 = bb[HW], d2 = bb[2 * HW], d3 = bb[3 * HW];
    float o  = obj[idx];

    float cx = (xx + 0.5f) * stride + d0 * stride;
    float cy = (yy + 0.5f) * stride + d1 * stride;
    float bw = __expf(d2) * stride;
    float bh = __expf(d3) * stride;
    float x1 = fminf(fmaxf(cx - bw * 0.5f, 0.f), img);
    float y1 = fminf(fmaxf(cy - bh * 0.5f, 0.f), img);
    float x2 = fminf(fmaxf(cx + bw * 0.5f, 0.f), img);
    float y2 = fminf(fmaxf(cy + bh * 0.5f, 0.f), img);

    boxes[(size_t)idx * 4 + 0] = x1;
    boxes[(size_t)idx * 4 + 1] = y1;
    boxes[(size_t)idx * 4 + 2] = x2;
    boxes[(size_t)idx * 4 + 3] = y2;

    float sc = o * maxc;
    scores[idx] = sc;
    maskf[idx]  = (o > 0.5f && sc > 0.3f) ? 1.f : 0.f;
    if (sp == 0) labels[b] = (float)label;
}

// ---------------------------------------------------------------------------
// Host-side orchestration
// ---------------------------------------------------------------------------
struct CB { const float *w, *b, *s, *o; };

static inline void launch_conv(hipStream_t stream,
                               const float* x, const float* w,
                               const float* bias, const float* bs, const float* bo,
                               float* y, int N, int Cin, int IH, int IW,
                               int Cout, int KH, int KW, int stride, int pad, int act)
{
    int OH = (IH + 2 * pad - KH) / stride + 1;
    int OW = (IW + 2 * pad - KW) / stride + 1;
    int NOut = N * OH * OW;
    dim3 grid((NOut + BT - 1) / BT, (Cout + BT - 1) / BT);
    if (KH == 7)
        conv_wmma_kernel<7, 7><<<grid, 128, 0, stream>>>(x, w, bias, bs, bo, y,
            N, Cin, IH, IW, Cout, OH, OW, stride, pad, act);
    else if (KH == 3)
        conv_wmma_kernel<3, 3><<<grid, 128, 0, stream>>>(x, w, bias, bs, bo, y,
            N, Cin, IH, IW, Cout, OH, OW, stride, pad, act);
    else
        conv_wmma_kernel<1, 1><<<grid, 128, 0, stream>>>(x, w, bias, bs, bo, y,
            N, Cin, IH, IW, Cout, OH, OW, stride, pad, act);
}

static inline void launch_pool(hipStream_t stream, const float* in, float* out,
                               int NC, int IH, int IW)
{
    int OH = (IH + 2 - 3) / 2 + 1;   // k=3, s=2, p=1
    int OW = (IW + 2 - 3) / 2 + 1;
    int total = NC * OH * OW;
    maxpool3x3s2_kernel<<<(total + 255) / 256, 256, 0, stream>>>(in, out, NC, IH, IW, OH, OW);
}

extern "C" void kernel_launch(void* const* d_in, const int* in_sizes, int n_in,
                              void* d_out, int out_size, void* d_ws, size_t ws_size,
                              hipStream_t stream)
{
    (void)in_sizes; (void)n_in; (void)out_size; (void)ws_size;

    // ---- unpack inputs in setup_inputs() dict order ----
    int ii = 0;
    const float* X = (const float*)d_in[ii++];
    auto nextCB = [&]() {
        CB c;
        c.w = (const float*)d_in[ii++]; c.b = (const float*)d_in[ii++];
        c.s = (const float*)d_in[ii++]; c.o = (const float*)d_in[ii++];
        return c;
    };
    CB c1 = nextCB();
    CB c2a = nextCB(), c2b = nextCB();
    CB c3a = nextCB(), c3b = nextCB(), c3c = nextCB();
    CB c4a = nextCB(), c4b = nextCB(), c4c = nextCB();
    const float* fc1_w = (const float*)d_in[ii++];
    const float* fc1_b = (const float*)d_in[ii++];
    const float* fc2_w = (const float*)d_in[ii++];
    const float* fc2_b = (const float*)d_in[ii++];
    const float* l1_w = (const float*)d_in[ii++]; const float* l1_b = (const float*)d_in[ii++];
    const float* l2_w = (const float*)d_in[ii++]; const float* l2_b = (const float*)d_in[ii++];
    const float* l3_w = (const float*)d_in[ii++]; const float* l3_b = (const float*)d_in[ii++];
    const float* bb_w = (const float*)d_in[ii++]; const float* bb_b = (const float*)d_in[ii++];
    const float* ob_w = (const float*)d_in[ii++]; const float* ob_b = (const float*)d_in[ii++];

    // ---- workspace partition (floats) ----
    float* ws = (float*)d_ws;
    float* R0 = ws;                     // 67,108,864 floats (268 MB)
    float* R1 = ws + 67108864;          // 33,554,432 floats (134 MB)
    float* S  = ws + 100663296;         // heads region
    float* gapb  = S;                   // 16*512
    float* h1    = S + 8192;            // 16*256
    float* clsl  = S + 12288;           // 16*4 (pad to 64)
    float* z1    = S + 12352;           // 4,194,304
    float* z2    = z1 + 4194304;        // 2,097,152
    float* z3    = z2 + 2097152;        // 1,048,576
    float* bboxb = z3 + 1048576;        // 65,536
    float* objb  = bboxb + 65536;       // 16,384

    const int N = 16;

    // ---- backbone ----
    launch_conv(stream, X, c1.w, c1.b, c1.s, c1.o, R0, N, 3, 512, 512, 64, 7, 7, 2, 3, 1);
    launch_pool(stream, R0, R1, N * 64, 256, 256);                    // -> [16,64,128,128]

    launch_conv(stream, R1, c2a.w, c2a.b, c2a.s, c2a.o, R0, N, 64, 128, 128, 128, 3, 3, 1, 1, 1);
    launch_conv(stream, R0, c2b.w, c2b.b, c2b.s, c2b.o, R1, N, 128, 128, 128, 128, 3, 3, 1, 1, 1);
    launch_pool(stream, R1, R0, N * 128, 128, 128);                   // -> [16,128,64,64]

    launch_conv(stream, R0, c3a.w, c3a.b, c3a.s, c3a.o, R1, N, 128, 64, 64, 256, 3, 3, 1, 1, 1);
    launch_conv(stream, R1, c3b.w, c3b.b, c3b.s, c3b.o, R0, N, 256, 64, 64, 256, 3, 3, 1, 1, 1);
    launch_conv(stream, R0, c3c.w, c3c.b, c3c.s, c3c.o, R1, N, 256, 64, 64, 256, 3, 3, 1, 1, 1);
    launch_pool(stream, R1, R0, N * 256, 64, 64);                     // -> [16,256,32,32]

    launch_conv(stream, R0, c4a.w, c4a.b, c4a.s, c4a.o, R1, N, 256, 32, 32, 512, 3, 3, 1, 1, 1);
    launch_conv(stream, R1, c4b.w, c4b.b, c4b.s, c4b.o, R0, N, 512, 32, 32, 512, 3, 3, 1, 1, 1);
    launch_conv(stream, R0, c4c.w, c4c.b, c4c.s, c4c.o, R1, N, 512, 32, 32, 512, 3, 3, 1, 1, 1);
    float* feat = R1;                                                 // [16,512,32,32]

    // ---- classifier head ----
    gap_kernel<<<N * 512, 256, 0, stream>>>(feat, gapb, 32 * 32);
    fc_kernel<<<(N * 256 + 255) / 256, 256, 0, stream>>>(gapb, fc1_w, fc1_b, h1, N, 512, 256, 1);
    fc_kernel<<<1, 64, 0, stream>>>(h1, fc2_w, fc2_b, clsl, N, 256, 4, 0);

    // ---- localizer head ----
    launch_conv(stream, feat, l1_w, l1_b, nullptr, nullptr, z1, N, 512, 32, 32, 256, 3, 3, 1, 1, 1);
    launch_conv(stream, z1,   l2_w, l2_b, nullptr, nullptr, z2, N, 256, 32, 32, 128, 3, 3, 1, 1, 1);
    launch_conv(stream, z2,   l3_w, l3_b, nullptr, nullptr, z3, N, 128, 32, 32,  64, 3, 3, 1, 1, 1);
    launch_conv(stream, z3,   bb_w, bb_b, nullptr, nullptr, bboxb, N, 64, 32, 32, 4, 1, 1, 1, 0, 0);
    launch_conv(stream, z3,   ob_w, ob_b, nullptr, nullptr, objb,  N, 64, 32, 32, 1, 1, 1, 1, 0, 2);

    // ---- decode to outputs: boxes | scores | labels | mask ----
    float* out      = (float*)d_out;
    float* o_boxes  = out;                 // 65536
    float* o_scores = out + 65536;         // 16384
    float* o_labels = out + 81920;         // 16
    float* o_mask   = out + 81936;         // 16384
    int total = N * 32 * 32;
    decode_kernel<<<(total + 255) / 256, 256, 0, stream>>>(
        bboxb, objb, clsl, o_boxes, o_scores, o_labels, o_mask,
        N, 32, 32, 512.0f, 16.0f);
}
